// SparseMOE_6889127542920
// MI455X (gfx1250) — matrix operand order repaired and checked by reading, not compiled, not run on hardware
//
#include <hip/hip_runtime.h>
#include <math.h>

// ---------------------------------------------------------------------------
// Sparse MoE (NoisyTop2) for gfx1250: bf16 WMMA compute, f32 accumulate,
// TDM (tensor_load_to_lds) staging of activation tiles, double-buffered LDS.
//   B=4, S=2048 -> N=8192 tokens, D=1024, E=8, K=2, H=4096
// out = sum_e (gate_e * relu(x@W1_e+b1_e)) @ W2_e + sum_e gate_e*b2_e
// ---------------------------------------------------------------------------

typedef __bf16 bf16_t;
typedef bf16_t v16bf __attribute__((ext_vector_type(16)));
typedef bf16_t v8bf  __attribute__((ext_vector_type(8)));
typedef float  v8f   __attribute__((ext_vector_type(8)));
typedef unsigned int u32x4 __attribute__((ext_vector_type(4)));
typedef int          i32x8 __attribute__((ext_vector_type(8)));
typedef int          i32x4 __attribute__((ext_vector_type(4)));

#define N_TOK 8192
#define DDIM  1024
#define EEXP  8
#define HDIM  4096
#define ATILE (128 * 32)   // elements per LDS tile

// --------------------------- TDM plumbing ----------------------------------
#if __has_builtin(__builtin_amdgcn_tensor_load_to_lds)
#define HAVE_TDM 1
#if __has_include(<hip/amd_detail/amd_gfx1250_TDM.h>)
// amdgpu-toolchain (clang-23): 6-arg form
#define TDM_ISSUE(g0, g1)                                                   \
    do {                                                                    \
        i32x4 _z4 = {0, 0, 0, 0};                                           \
        i32x8 _z8 = {0, 0, 0, 0, 0, 0, 0, 0};                               \
        __builtin_amdgcn_tensor_load_to_lds((g0), (g1), _z4, _z4, _z8, 0);  \
    } while (0)
#else
// ROCm 7.2 (clang-22): 5-arg form
#define TDM_ISSUE(g0, g1)                                                   \
    do {                                                                    \
        i32x4 _z4 = {0, 0, 0, 0};                                           \
        __builtin_amdgcn_tensor_load_to_lds((g0), (g1), _z4, _z4, 0);       \
    } while (0)
#endif

// 2D tile load: tile_d0 x tile_d1 elements of 2 bytes, row pitch = stride0
// elements. Descriptor bit layout per CDNA5 ISA 8.3 / 8.4.
__device__ __forceinline__ void tdm_load_tile_bf16(
    unsigned int lds_off, const bf16_t* gptr,
    unsigned int tensor_d0, unsigned int tensor_d1,
    unsigned int tile_d0, unsigned int tile_d1, unsigned int stride0)
{
    unsigned long long ga = (unsigned long long)(uintptr_t)gptr;
    u32x4 g0;
    g0[0] = 1u;                                            // count=1, user mode
    g0[1] = lds_off;                                       // lds_addr
    g0[2] = (unsigned int)ga;                              // global_addr[31:0]
    g0[3] = ((unsigned int)(ga >> 32) & 0x01FFFFFFu)       // global_addr[56:32]
            | 0x80000000u;                                 // type=2 (image)
    i32x8 g1;
    g1[0] = (int)(1u << 16);                               // data_size=1 (2B)
    g1[1] = (int)((tensor_d0 & 0xFFFFu) << 16);            // dim0[15:0] @ b63:48
    g1[2] = (int)((tensor_d0 >> 16) | ((tensor_d1 & 0xFFFFu) << 16));
    g1[3] = (int)((tensor_d1 >> 16) | (tile_d0 << 16));    // tile_dim0 @ b127:112
    g1[4] = (int)(tile_d1);                                // tile_dim1, tile_dim2=0
    g1[5] = (int)(stride0);                                // dim0_stride[31:0]
    g1[6] = 0;
    g1[7] = 0;
    TDM_ISSUE(g0, g1);
}
#endif

// ---------------------------------------------------------------------------
// Router: one wave32 per token.
// ---------------------------------------------------------------------------
__global__ __launch_bounds__(256) void moe_router(
    const float* __restrict__ x, const float* __restrict__ noise,
    const float* __restrict__ Wg, const float* __restrict__ bg,
    const float* __restrict__ Wn, const float* __restrict__ bn,
    float* __restrict__ gate)
{
    int wave = threadIdx.x >> 5;
    int lane = threadIdx.x & 31;
    int token = blockIdx.x * 8 + wave;
    if (token >= N_TOK) return;

    float ag[EEXP], an[EEXP];
#pragma unroll
    for (int e = 0; e < EEXP; ++e) { ag[e] = 0.f; an[e] = 0.f; }

    const float* xr = x + (size_t)token * DDIM;
    for (int d = lane; d < DDIM; d += 32) {
        float xv = xr[d];
        const float* wgr = Wg + d * EEXP;
        const float* wnr = Wn + d * EEXP;
#pragma unroll
        for (int e = 0; e < EEXP; ++e) {
            ag[e] += xv * wgr[e];
            an[e] += xv * wnr[e];
        }
    }
#pragma unroll
    for (int e = 0; e < EEXP; ++e) {
#pragma unroll
        for (int off = 16; off > 0; off >>= 1) {
            ag[e] += __shfl_xor(ag[e], off, 32);
            an[e] += __shfl_xor(an[e], off, 32);
        }
    }
    if (lane == 0) {
        float nl[EEXP];
#pragma unroll
        for (int e = 0; e < EEXP; ++e) {
            float lg = ag[e] + bg[e];
            float sv = an[e] + bn[e];
            float sp = (sv > 20.f) ? sv : log1pf(expf(sv));   // softplus
            nl[e] = lg + noise[(size_t)token * EEXP + e] * sp;
        }
        // top-2 (first occurrence on ties, like lax.top_k)
        int i0 = 0;
#pragma unroll
        for (int e = 1; e < EEXP; ++e) if (nl[e] > nl[i0]) i0 = e;
        int i1 = -1;
#pragma unroll
        for (int e = 0; e < EEXP; ++e)
            if (e != i0 && (i1 < 0 || nl[e] > nl[i1])) i1 = e;
        float m  = nl[i0];
        float e0 = expf(nl[i0] - m);
        float e1 = expf(nl[i1] - m);
        float iZ = 1.f / (e0 + e1);
#pragma unroll
        for (int e = 0; e < EEXP; ++e) {
            float g = (e == i0) ? e0 * iZ : ((e == i1) ? e1 * iZ : 0.f);
            gate[(size_t)token * EEXP + e] = g;
        }
    }
}

// ---------------------------------------------------------------------------
// x (f32) -> xb (bf16), vectorized
// ---------------------------------------------------------------------------
__global__ __launch_bounds__(256) void moe_cvt_x(
    const float* __restrict__ x, bf16_t* __restrict__ xb)
{
    size_t i = ((size_t)blockIdx.x * blockDim.x + threadIdx.x) * 4;
    if (i >= (size_t)N_TOK * DDIM) return;
    float4 v = *(const float4*)(x + i);
    bf16_t o[4] = { (bf16_t)v.x, (bf16_t)v.y, (bf16_t)v.z, (bf16_t)v.w };
    *(uint2*)(xb + i) = *(const uint2*)o;
}

// ---------------------------------------------------------------------------
// out init: out[n,d] = sum_e gate[n,e] * b2[e,d]
// ---------------------------------------------------------------------------
__global__ __launch_bounds__(256) void moe_init_out(
    const float* __restrict__ gate, const float* __restrict__ b2,
    float* __restrict__ out)
{
    size_t i = (size_t)blockIdx.x * blockDim.x + threadIdx.x;
    if (i >= (size_t)N_TOK * DDIM) return;
    int token = (int)(i / DDIM);
    int dcol  = (int)(i % DDIM);
    float acc = 0.f;
#pragma unroll
    for (int e = 0; e < EEXP; ++e)
        acc += gate[(size_t)token * EEXP + e] * b2[(size_t)e * DDIM + dcol];
    out[i] = acc;
}

// ---------------------------------------------------------------------------
// Fragment helpers matching CDNA5 WMMA VGPR layouts (wave32).
// A 16x32 bf16: lane half h, row r: elements 0..7 -> K = 8h..8h+7,
//               elements 8..15 -> K = 16+8h..16+8h+7.
// B 32x16 bf16: lane half h, col c: elements 0..7 -> K = 8h..8h+7,
//               elements 8..15 -> K = 16+8h..16+8h+7.
// C/D 16x16 f32: lane half h, col r: element i -> row i+8h.
// ---------------------------------------------------------------------------
__device__ __forceinline__ v16bf load_a_frag(const bf16_t* As, int row, int half) {
    v8bf lo = *(const v8bf*)(As + row * 32 + half * 8);
    v8bf hi = *(const v8bf*)(As + row * 32 + 16 + half * 8);
    return __builtin_shufflevector(lo, hi, 0,1,2,3,4,5,6,7,8,9,10,11,12,13,14,15);
}
__device__ __forceinline__ v16bf load_b_frag(const bf16_t* Bs, int col, int half) {
    v8bf lo = *(const v8bf*)(Bs + col * 32 + half * 8);
    v8bf hi = *(const v8bf*)(Bs + col * 32 + 16 + half * 8);
    return __builtin_shufflevector(lo, hi, 0,1,2,3,4,5,6,7,8,9,10,11,12,13,14,15);
}

// ---------------------------------------------------------------------------
// Shared GEMM mainloop (double-buffered, TDM lookahead):
//   acc(128x128 block) += A[mBase:, k] * bf16(W[k, nBase:])
//   A:  bf16 row-major [*, lda], DMA'd to LDS by TDM (wave 0 issues)
//   W:  f32  row-major [KDIM, ldb], 4x4 micro-transposed into LDS [col][k]
// ---------------------------------------------------------------------------
__device__ __forceinline__ void gemm_mainloop(
    v8f acc[4][2],
    const bf16_t* __restrict__ A, int lda, int KDIM,
    const float* __restrict__ W, int ldb,
    bf16_t* As, bf16_t* Bs,
    int mBase, int nBase, int tid, int wave, int half, int r, int wm, int wn)
{
#if defined(HAVE_TDM)
    const unsigned int As_lds = (unsigned int)(uintptr_t)As;
#endif
    const int c4 = (tid & 31) * 4;   // B stage: this thread's 4 columns
    const int k4 = (tid >> 5) * 4;   // B stage: this thread's 4 k-rows

    // stage A tile for K-chunk kc into LDS buffer
    auto stage_A = [&](int kc, int buf) {
#if defined(HAVE_TDM)
        if (wave == 0)
            tdm_load_tile_bf16(As_lds + (unsigned int)(buf * ATILE * 2),
                               A + (size_t)mBase * lda + kc,
                               (unsigned int)lda, (unsigned int)N_TOK,
                               32u, 128u, (unsigned int)lda);
#else
        bf16_t* Asd = As + buf * ATILE;
#pragma unroll
        for (int i = 0; i < 2; ++i) {
            int li = tid + i * 256;
            int row = li >> 2, seg = li & 3;
            const uint4* src = (const uint4*)(A + (size_t)(mBase + row) * lda + kc + seg * 8);
            *(uint4*)(Asd + row * 32 + seg * 8) = *src;
        }
#endif
    };
    // stage B tile (32 k-rows x 128 cols f32) into LDS [col][k] bf16:
    // each thread does a 4x4 register micro-transpose.
    auto stage_B = [&](int kc, int buf) {
        bf16_t* Bsd = Bs + buf * ATILE;
        const float* src = W + (size_t)(kc + k4) * ldb + nBase + c4;
        if (kc + 32 < KDIM)
            __builtin_prefetch(src + (size_t)32 * ldb, 0, 1);
        float4 r0 = *(const float4*)(src);
        float4 r1 = *(const float4*)(src + ldb);
        float4 r2 = *(const float4*)(src + 2 * (size_t)ldb);
        float4 r3 = *(const float4*)(src + 3 * (size_t)ldb);
        bf16_t t0[4] = { (bf16_t)r0.x, (bf16_t)r1.x, (bf16_t)r2.x, (bf16_t)r3.x };
        bf16_t t1[4] = { (bf16_t)r0.y, (bf16_t)r1.y, (bf16_t)r2.y, (bf16_t)r3.y };
        bf16_t t2[4] = { (bf16_t)r0.z, (bf16_t)r1.z, (bf16_t)r2.z, (bf16_t)r3.z };
        bf16_t t3[4] = { (bf16_t)r0.w, (bf16_t)r1.w, (bf16_t)r2.w, (bf16_t)r3.w };
        *(uint2*)(Bsd + (c4 + 0) * 32 + k4) = *(const uint2*)t0;
        *(uint2*)(Bsd + (c4 + 1) * 32 + k4) = *(const uint2*)t1;
        *(uint2*)(Bsd + (c4 + 2) * 32 + k4) = *(const uint2*)t2;
        *(uint2*)(Bsd + (c4 + 3) * 32 + k4) = *(const uint2*)t3;
    };

    // prologue: fill buffer 0
    stage_A(0, 0);
    stage_B(0, 0);

    for (int k0 = 0; k0 < KDIM; k0 += 32) {
        const int buf = (k0 >> 5) & 1;
#if defined(HAVE_TDM)
        if (wave == 0)
            __builtin_amdgcn_s_wait_tensorcnt((short)0);  // current A tile landed
#endif
        __syncthreads();   // publish current tiles; prior readers of next buf done

        if (k0 + 32 < KDIM) {       // lookahead stage into the other buffer
            stage_A(k0 + 32, buf ^ 1);
            stage_B(k0 + 32, buf ^ 1);
        }

        const bf16_t* Asb = As + buf * ATILE;
        const bf16_t* Bsb = Bs + buf * ATILE;
        v16bf afrag[4], bfrag[2];
#pragma unroll
        for (int mi = 0; mi < 4; ++mi)
            afrag[mi] = load_a_frag(Asb, wm * 64 + mi * 16 + r, half);
#pragma unroll
        for (int nj = 0; nj < 2; ++nj)
            bfrag[nj] = load_b_frag(Bsb, wn * 32 + nj * 16 + r, half);

#pragma unroll
        for (int mi = 0; mi < 4; ++mi)
#pragma unroll
            for (int nj = 0; nj < 2; ++nj)
                acc[mi][nj] = __builtin_amdgcn_wmma_f32_16x16x32_bf16(
                    false, afrag[mi], false, bfrag[nj],
                    (short)0, acc[mi][nj], false, false);
    }
}

// ---------------------------------------------------------------------------
// GEMM1: Hg[n,h] = gate[n,e] * relu( xb[n,:] @ W1_e[:,h] + b1_e[h] ), bf16 out.
// Block tile 128x128, K-chunk 32, 8 waves (2x4), wave tile 64x32.
// ---------------------------------------------------------------------------
__global__ __launch_bounds__(256) void moe_gemm1(
    const bf16_t* __restrict__ xb,       // [N_TOK, DDIM] bf16
    const float*  __restrict__ W1e,      // [DDIM, HDIM] f32 (expert slice)
    const float*  __restrict__ b1e,      // [HDIM]
    const float*  __restrict__ gate,     // [N_TOK, EEXP]
    int e,
    bf16_t* __restrict__ Hg)             // [N_TOK, HDIM] bf16
{
    __shared__ __align__(16) bf16_t As[2 * ATILE];
    __shared__ __align__(16) bf16_t Bs[2 * ATILE];
    __shared__ float gate_s[128];
    __shared__ float bias_s[128];

    const int tid  = threadIdx.x;
    const int wave = tid >> 5, lane = tid & 31;
    const int wm = wave >> 2, wn = wave & 3;
    const int half = lane >> 4, r = lane & 15;
    const int mBase = blockIdx.y * 128;
    const int nBase = blockIdx.x * 128;

    if (tid < 128)       gate_s[tid] = gate[(size_t)(mBase + tid) * EEXP + e];
    else                 bias_s[tid - 128] = b1e[nBase + (tid - 128)];

    v8f acc[4][2] = {};
    gemm_mainloop(acc, xb, DDIM, DDIM, W1e, HDIM, As, Bs,
                  mBase, nBase, tid, wave, half, r, wm, wn);

    // epilogue: bias + relu + gate, store bf16
#pragma unroll
    for (int mi = 0; mi < 4; ++mi) {
#pragma unroll
        for (int nj = 0; nj < 2; ++nj) {
            int colL = wn * 32 + nj * 16 + r;
            float bv = bias_s[colL];
#pragma unroll
            for (int i = 0; i < 8; ++i) {
                int rowL = wm * 64 + mi * 16 + 8 * half + i;
                float v = acc[mi][nj][i] + bv;
                v = v > 0.f ? v : 0.f;
                v *= gate_s[rowL];
                Hg[(size_t)(mBase + rowL) * HDIM + nBase + colL] = (bf16_t)v;
            }
        }
    }
}

// ---------------------------------------------------------------------------
// GEMM2: out[n,d] += Hg[n,:] @ W2_e[:,d]
// ---------------------------------------------------------------------------
__global__ __launch_bounds__(256) void moe_gemm2(
    const bf16_t* __restrict__ Hg,       // [N_TOK, HDIM] bf16
    const float*  __restrict__ W2e,      // [HDIM, DDIM] f32 (expert slice)
    float* __restrict__ out)             // [N_TOK, DDIM] f32 (accumulated)
{
    __shared__ __align__(16) bf16_t As[2 * ATILE];
    __shared__ __align__(16) bf16_t Bs[2 * ATILE];

    const int tid  = threadIdx.x;
    const int wave = tid >> 5, lane = tid & 31;
    const int wm = wave >> 2, wn = wave & 3;
    const int half = lane >> 4, r = lane & 15;
    const int mBase = blockIdx.y * 128;
    const int nBase = blockIdx.x * 128;

    v8f acc[4][2] = {};
    gemm_mainloop(acc, Hg, HDIM, HDIM, W2e, DDIM, As, Bs,
                  mBase, nBase, tid, wave, half, r, wm, wn);

#pragma unroll
    for (int mi = 0; mi < 4; ++mi) {
#pragma unroll
        for (int nj = 0; nj < 2; ++nj) {
            int colL = wn * 32 + nj * 16 + r;
#pragma unroll
            for (int i = 0; i < 8; ++i) {
                int rowL = wm * 64 + mi * 16 + 8 * half + i;
                size_t idx = (size_t)(mBase + rowL) * DDIM + nBase + colL;
                out[idx] = out[idx] + acc[mi][nj][i];   // tile exclusively owned
            }
        }
    }
}

// ---------------------------------------------------------------------------
// Launch
// ---------------------------------------------------------------------------
extern "C" void kernel_launch(void* const* d_in, const int* in_sizes, int n_in,
                              void* d_out, int out_size, void* d_ws, size_t ws_size,
                              hipStream_t stream)
{
    const float* x     = (const float*)d_in[0];
    const float* noise = (const float*)d_in[1];
    const float* Wg    = (const float*)d_in[2];
    const float* bg    = (const float*)d_in[3];
    const float* Wn    = (const float*)d_in[4];
    const float* bn    = (const float*)d_in[5];
    const float* W1    = (const float*)d_in[6];
    const float* b1    = (const float*)d_in[7];
    const float* W2    = (const float*)d_in[8];
    const float* b2    = (const float*)d_in[9];
    float* out = (float*)d_out;

    // workspace layout
    char* ws = (char*)d_ws;
    float*  gate = (float*)ws;                                  // N*E*4   = 256 KB
    size_t  off  = ((size_t)N_TOK * EEXP * sizeof(float) + 255) & ~(size_t)255;
    bf16_t* xb   = (bf16_t*)(ws + off);                         // N*D*2   = 16 MB
    off += ((size_t)N_TOK * DDIM * sizeof(bf16_t) + 255) & ~(size_t)255;
    bf16_t* Hg   = (bf16_t*)(ws + off);                         // N*H*2   = 67 MB

    // 1) router
    moe_router<<<N_TOK / 8, 256, 0, stream>>>(x, noise, Wg, bg, Wn, bn, gate);
    // 2) x -> bf16
    {
        size_t n = (size_t)N_TOK * DDIM / 4;
        moe_cvt_x<<<(unsigned)((n + 255) / 256), 256, 0, stream>>>(x, xb);
    }
    // 3) out = sum_e gate_e * b2_e
    {
        size_t n = (size_t)N_TOK * DDIM;
        moe_init_out<<<(unsigned)((n + 255) / 256), 256, 0, stream>>>(gate, b2, out);
    }
    // 4) per-expert GEMMs
    for (int e = 0; e < EEXP; ++e) {
        const float* W1e = W1 + (size_t)e * DDIM * HDIM;
        const float* b1e = b1 + (size_t)e * HDIM;
        const float* W2e = W2 + (size_t)e * HDIM * DDIM;
        dim3 g1(HDIM / 128, N_TOK / 128);
        moe_gemm1<<<g1, 256, 0, stream>>>(xb, W1e, b1e, gate, e, Hg);
        dim3 g2(DDIM / 128, N_TOK / 128);
        moe_gemm2<<<g2, 256, 0, stream>>>(Hg, W2e, out);
    }
}
